// DepthNet_43069932044809
// MI455X (gfx1250) — matrix operand order repaired and checked by reading, compiled, not run on hardware
//
#include <hip/hip_runtime.h>
#include <hip/hip_bf16.h>
#include <math.h>

// Problem constants (from reference): V,B,C,H,W,D,G = 5,2,32,128,160,32,8
#define V_ 5
#define B_ 2
#define C_ 32
#define H_ 128
#define W_ 160
#define D_ 32
#define G_ 8
#define HW_ (H_*W_)

// d_out layout (floats, concatenated in reference return order)
#define OUT_DEPTH 0
#define OUT_CONF  (B_*HW_)
#define OUT_VW    (2*B_*HW_)
#define OUT_PVP   (2*B_*HW_ + B_*4*HW_)
#define OUT_SIM   (OUT_PVP + B_*D_*HW_)

// workspace layout (floats). Requires ~42.3 MB of d_ws.
#define WS_PROJ 0
#define WS_BIG  128                                   // sim_tmp, later reused as conv1 activations
#define WS_WSUM (128 + (size_t)B_*G_*D_*HW_)

typedef __attribute__((ext_vector_type(16))) _Float16 v16h;
typedef __attribute__((ext_vector_type(8)))  float    v8f;

// ---------------------------------------------------------------- projections
__device__ inline void combine34(const float* pm, int b, int v, float Cm[3][4]) {
  // pm layout (B, V, 2, 4, 4); top 3x4 of result = K[:3,:3] @ ext[:3,:4]
  const float* ext = pm + (((size_t)b*V_ + v)*2 + 0)*16;
  const float* K   = pm + (((size_t)b*V_ + v)*2 + 1)*16;
  for (int i = 0; i < 3; ++i)
    for (int j = 0; j < 4; ++j) {
      float s = 0.f;
      for (int l = 0; l < 3; ++l) s += K[i*4+l]*ext[l*4+j];
      Cm[i][j] = s;
    }
}

__global__ void setup_proj_kernel(const float* __restrict__ pm, float* __restrict__ wsproj) {
  int t = threadIdx.x;
  if (t >= B_*(V_-1)) return;
  int b = t / (V_-1);
  int v = 1 + t % (V_-1);
  // reference projection (affine: bottom row of combined matrix is [0,0,0,1])
  float R0[3][4]; combine34(pm, b, 0, R0);
  float a00=R0[0][0],a01=R0[0][1],a02=R0[0][2];
  float a10=R0[1][0],a11=R0[1][1],a12=R0[1][2];
  float a20=R0[2][0],a21=R0[2][1],a22=R0[2][2];
  float det = a00*(a11*a22-a12*a21) - a01*(a10*a22-a12*a20) + a02*(a10*a21-a11*a20);
  float id = 1.0f/det;
  float Ri[3][3];
  Ri[0][0]= (a11*a22-a12*a21)*id; Ri[0][1]=-(a01*a22-a02*a21)*id; Ri[0][2]= (a01*a12-a02*a11)*id;
  Ri[1][0]=-(a10*a22-a12*a20)*id; Ri[1][1]= (a00*a22-a02*a20)*id; Ri[1][2]=-(a00*a12-a02*a10)*id;
  Ri[2][0]= (a10*a21-a11*a20)*id; Ri[2][1]=-(a00*a21-a01*a20)*id; Ri[2][2]= (a00*a11-a01*a10)*id;
  float ti[3];
  for (int i = 0; i < 3; ++i)
    ti[i] = -(Ri[i][0]*R0[0][3] + Ri[i][1]*R0[1][3] + Ri[i][2]*R0[2][3]);
  float Cv[3][4]; combine34(pm, b, v, Cv);
  float* o = wsproj + ((size_t)b*(V_-1) + (v-1))*12;
  for (int i = 0; i < 3; ++i) {
    for (int j = 0; j < 3; ++j) {
      float s = 0.f;
      for (int l = 0; l < 3; ++l) s += Cv[i][l]*Ri[l][j];
      o[i*3+j] = s;
    }
    float s = Cv[i][3];
    for (int l = 0; l < 3; ++l) s += Cv[i][l]*ti[l];
    o[9+i] = s;
  }
}

// ---------------------------------------------------------------- warp + group similarity
__global__ void warp_sim_kernel(const float* __restrict__ feat,
                                const float* __restrict__ depthv,
                                const float* __restrict__ wsproj,
                                float* __restrict__ sim_tmp, int v) {
  long idx = (long)blockIdx.x*blockDim.x + threadIdx.x;
  if (idx >= (long)B_*D_*HW_) return;
  int hw = (int)(idx % HW_);
  int d  = (int)((idx / HW_) % D_);
  int b  = (int)(idx / ((long)D_*HW_));
  int h = hw / W_, w = hw % W_;
  const float* P = wsproj + ((size_t)b*(V_-1) + (v-1))*12;
  float x = (float)w, y = (float)h;
  float rx = P[0]*x + P[1]*y + P[2];
  float ry = P[3]*x + P[4]*y + P[5];
  float rz = P[6]*x + P[7]*y + P[8];
  float dv = depthv[((size_t)b*D_ + d)*HW_ + hw];
  float pz = rz*dv + P[11];
  float inv = 1.0f/pz;
  float px = (rx*dv + P[9])*inv;
  float py = (ry*dv + P[10])*inv;
  float x0 = floorf(px), y0 = floorf(py);
  float wx1 = px - x0, wy1 = py - y0;
  float wxv[2] = {1.f - wx1, wx1};
  float wyv[2] = {1.f - wy1, wy1};
  float xs[2]  = {x0, x0 + 1.f};
  float ysv[2] = {y0, y0 + 1.f};
  float cw[4]; int co[4]; int c = 0;
  for (int jy = 0; jy < 2; ++jy)
    for (int jx = 0; jx < 2; ++jx) {
      float xi = xs[jx], yi = ysv[jy];
      float valid = (xi >= 0.f && xi <= (float)(W_-1) && yi >= 0.f && yi <= (float)(H_-1)) ? 1.f : 0.f;
      int xc = (int)fminf(fmaxf(xi, 0.f), (float)(W_-1));
      int yc = (int)fminf(fmaxf(yi, 0.f), (float)(H_-1));
      cw[c] = wxv[jx]*wyv[jy]*valid;
      co[c] = yc*W_ + xc;
      ++c;
    }
  const float* fsrc = feat + ((size_t)v*B_ + b)*C_*HW_;
  const float* fref = feat + ((size_t)b)*C_*HW_;   // view 0
  for (int g = 0; g < G_; ++g) {
    float acc = 0.f;
    for (int cc = 0; cc < C_/G_; ++cc) {
      int ch = g*(C_/G_) + cc;
      const float* fc = fsrc + (size_t)ch*HW_;
      float wv = fc[co[0]]*cw[0] + fc[co[1]]*cw[1] + fc[co[2]]*cw[2] + fc[co[3]]*cw[3];
      acc += wv * fref[(size_t)ch*HW_ + hw];
    }
    sim_tmp[(((size_t)b*G_ + g)*D_ + d)*HW_ + hw] = acc * (1.0f/(float)(C_/G_));
  }
}

// ---------------------------------------------------------------- entropy -> MLP -> view weight
__global__ void view_weight_kernel(const float* __restrict__ sim_tmp,
                                   const float* __restrict__ pw1, const float* __restrict__ pb1,
                                   const float* __restrict__ pw2, const float* __restrict__ pb2,
                                   const float* __restrict__ pw3, const float* __restrict__ pb3,
                                   float* __restrict__ out_vw, float* __restrict__ wsum,
                                   int v, int init) {
  long idx = (long)blockIdx.x*blockDim.x + threadIdx.x;
  if (idx >= (long)B_*HW_) return;
  int hw = (int)(idx % HW_);
  int b  = (int)(idx / HW_);
  float w1l[16], b1l[16], w2l[128], b2l[8], w3l[8];
  for (int i = 0; i < 16; ++i) { w1l[i] = pw1[i]; b1l[i] = pb1[i]; }
  for (int i = 0; i < 128; ++i) w2l[i] = pw2[i];
  for (int i = 0; i < 8; ++i)  { b2l[i] = pb2[i]; w3l[i] = pw3[i]; }
  float b3l = pb3[0];
  float best = -1e30f;
  for (int d = 0; d < D_; ++d) {
    float s[G_]; float m = -1e30f;
    for (int g = 0; g < G_; ++g) {
      s[g] = sim_tmp[(((size_t)b*G_ + g)*D_ + d)*HW_ + hw];
      m = fmaxf(m, s[g]);
    }
    float sum = 0.f;
    for (int g = 0; g < G_; ++g) { s[g] = expf(s[g]-m); sum += s[g]; }
    float isum = 1.f/sum, ent = 0.f;
    for (int g = 0; g < G_; ++g) { float p = s[g]*isum; ent -= p*logf(p + 1e-7f); }
    float h1[16];
    for (int i = 0; i < 16; ++i) h1[i] = fmaxf(w1l[i]*ent + b1l[i], 0.f);
    float o = b3l;
    for (int j = 0; j < 8; ++j) {
      float a = b2l[j];
      for (int i = 0; i < 16; ++i) a += w2l[j*16+i]*h1[i];
      o += w3l[j]*fmaxf(a, 0.f);
    }
    best = fmaxf(best, 1.f/(1.f + expf(-o)));
  }
  out_vw[((size_t)b*4 + (v-1))*HW_ + hw] = best;
  if (init) wsum[idx] = best; else wsum[idx] += best;
}

// ---------------------------------------------------------------- weighted accumulation
__global__ void accum_sim_kernel(const float* __restrict__ sim_tmp,
                                 const float* __restrict__ vw_all,
                                 float* __restrict__ sim_out, int v, int init) {
  long idx = (long)blockIdx.x*blockDim.x + threadIdx.x;
  if (idx >= (long)B_*G_*D_*HW_) return;
  int hw = (int)(idx % HW_);
  int b  = (int)(idx / ((long)G_*D_*HW_));
  float vw = vw_all[((size_t)b*4 + (v-1))*HW_ + hw];
  float val = sim_tmp[idx]*vw;
  sim_out[idx] = init ? val : (sim_out[idx] + val);
}

__global__ void norm_sim_kernel(float* __restrict__ sim_out, const float* __restrict__ wsum) {
  long idx = (long)blockIdx.x*blockDim.x + threadIdx.x;
  if (idx >= (long)B_*G_*D_*HW_) return;
  int hw = (int)(idx % HW_);
  int b  = (int)(idx / ((long)G_*D_*HW_));
  sim_out[idx] = sim_out[idx] / (wsum[(size_t)b*HW_ + hw] + 1e-6f);
}

// ---------------------------------------------------------------- conv1 as implicit GEMM on WMMA
// Out[m,n] over K = ic(8) x 27 taps = 216, padded to 7 chunks of K=32.
// M = 16 consecutive W positions, N = 16 (8 real out channels).
// One block = 5 waves = half a W-row (5 tiles of 16 = 80 outputs), input staged in LDS as f16.
#define KTOT 216
#define NCHUNK 7
#define CV1_WAVES 5
#define CV1_THREADS (CV1_WAVES*32)
#define TILE_X 82                       // 80 outputs + 2 halo
#define NROWS 72                        // 8 ic * 3 kd * 3 kh
#define ZROW NROWS                      // extra all-zero row for K-padding
#define LDSA_SIZE ((NROWS+1)*TILE_X)    // 5986 halves
#define CV1_BLOCKS (B_*D_*H_*2)         // 2 half-rows per (b,d,h)

__global__ __launch_bounds__(CV1_THREADS) void conv1_wmma_kernel(
    const float* __restrict__ simf, const float* __restrict__ w1,
    const float* __restrict__ b1, float* __restrict__ h1out) {
  // weight fragments in B-matrix per-lane layout: lane=(n=lane%16, hi=lane/16), elem e -> K=16*hi+e
  __shared__ __align__(32) _Float16 wlds[NCHUNK][32][16];
  __shared__ __align__(32) _Float16 alds[LDSA_SIZE];
  __shared__ int kTab[NCHUNK*32];       // k -> row*TILE_X + kw (zero row for k >= KTOT)

  int tid = threadIdx.x;

  for (int pos = tid; pos < NCHUNK*32*16; pos += CV1_THREADS) {
    int chunk = pos >> 9;
    int rem   = pos & 511;
    int lane  = rem >> 4;
    int e     = rem & 15;
    int n  = lane & 15;
    int hi = lane >> 4;
    int k  = chunk*32 + 16*hi + e;
    float val = (n < 8 && k < KTOT) ? w1[n*KTOT + k] : 0.0f;
    wlds[chunk][lane][e] = (_Float16)val;
  }
  for (int k = tid; k < NCHUNK*32; k += CV1_THREADS) {
    int off;
    if (k < KTOT) {
      int ic = k/27;  int rr = k - ic*27;
      int kd = rr/9;  rr -= kd*9;
      int kh = rr/3;  int kw = rr - kh*3;
      off = (ic*9 + kd*3 + kh)*TILE_X + kw;
    } else {
      off = ZROW*TILE_X;
    }
    kTab[k] = off;
  }

  // block -> (b, d, h, half-row)
  int blk  = blockIdx.x;
  int half = blk & 1;
  int r1   = blk >> 1;
  int h = r1 % H_; r1 /= H_;
  int d = r1 % D_;
  int b = r1 / D_;
  int wbase = half*80;                  // output w in [wbase, wbase+80)

  // stage input tile (coalesced along w), zero-padded halo, f16
  for (int pos = tid; pos < LDSA_SIZE; pos += CV1_THREADS) {
    int row = pos / TILE_X;
    int x   = pos - row*TILE_X;
    float val = 0.0f;
    if (row < NROWS) {
      int ic = row/9;  int rr = row - ic*9;
      int kd = rr/3;   int kh = rr - kd*3;
      int zd = d + kd - 1, zh = h + kh - 1;
      int zw = wbase + x - 1;
      if (zd >= 0 && zd < D_ && zh >= 0 && zh < H_ && zw >= 0 && zw < W_)
        val = simf[(((size_t)b*G_ + ic)*D_ + zd)*HW_ + zh*W_ + zw];
    }
    alds[pos] = (_Float16)val;
  }
  __syncthreads();

  int wave = tid >> 5;                  // tile within block
  int lane = tid & 31;
  int m  = lane & 15;
  int hi = lane >> 4;
  int xbase = 16*wave + m;              // lds x-index base (kw comes from kTab)

  v8f acc = {0.f,0.f,0.f,0.f,0.f,0.f,0.f,0.f};
#pragma unroll
  for (int chunk = 0; chunk < NCHUNK; ++chunk) {
    v16h afrag;
#pragma unroll
    for (int e = 0; e < 16; ++e) {
      // A layout matching B fragments: elem e -> K=16*hi+e
      int kl = 16*hi + e;
      afrag[e] = alds[kTab[chunk*32 + kl] + xbase];    // branchless ds loads
    }
    v16h bfrag = *(const v16h*)(&wlds[chunk][lane][0]);
    acc = __builtin_amdgcn_wmma_f32_16x16x32_f16(false, afrag, false, bfrag,
                                                 (short)0, acc, false, false);
  }

  int n = lane & 15;
  if (n < 8) {
    float bias = b1[n];
    int w0 = wbase + 16*wave;
#pragma unroll
    for (int r = 0; r < 8; ++r) {
      int mm = r + 8*hi;                // C/D layout: vgpr r, lanes 16-31 -> M = r+8
      float o = fmaxf(acc[r] + bias, 0.0f);
      h1out[(((size_t)b*8 + n)*D_ + d)*HW_ + h*W_ + (w0 + mm)] = o;
    }
  }
}

// ---------------------------------------------------------------- conv2 (1 out channel) scalar
__global__ void conv2_kernel(const float* __restrict__ h1, const float* __restrict__ w2,
                             const float* __restrict__ b2, float* __restrict__ pvp) {
  long idx = (long)blockIdx.x*blockDim.x + threadIdx.x;
  if (idx >= (long)B_*D_*HW_) return;
  int hw = (int)(idx % HW_);
  int d  = (int)((idx / HW_) % D_);
  int b  = (int)(idx / ((long)D_*HW_));
  int h = hw / W_, w = hw % W_;
  float acc = b2[0];
  for (int ic = 0; ic < 8; ++ic)
    for (int kd = 0; kd < 3; ++kd) {
      int zd = d + kd - 1; if (zd < 0 || zd >= D_) continue;
      for (int kh = 0; kh < 3; ++kh) {
        int zh = h + kh - 1; if (zh < 0 || zh >= H_) continue;
        for (int kw = 0; kw < 3; ++kw) {
          int zw = w + kw - 1; if (zw < 0 || zw >= W_) continue;
          acc += h1[(((size_t)b*8 + ic)*D_ + zd)*HW_ + zh*W_ + zw] * w2[ic*27 + kd*9 + kh*3 + kw];
        }
      }
    }
  pvp[idx] = acc;
}

// ---------------------------------------------------------------- softmax-over-depth, depth, conf
__global__ void depth_kernel(const float* __restrict__ pvp, const float* __restrict__ depthv,
                             float* __restrict__ dout, float* __restrict__ conf) {
  long idx = (long)blockIdx.x*blockDim.x + threadIdx.x;
  if (idx >= (long)B_*HW_) return;
  int hw = (int)(idx % HW_);
  int b  = (int)(idx / HW_);
  float pv[D_];
  float m = -1e30f;
  for (int d = 0; d < D_; ++d) { pv[d] = pvp[((size_t)b*D_ + d)*HW_ + hw]; m = fmaxf(m, pv[d]); }
  float s = 0.f;
  for (int d = 0; d < D_; ++d) { pv[d] = expf(pv[d]-m); s += pv[d]; }
  float isum = 1.f/s;
  float depth = 0.f, di = 0.f;
  for (int d = 0; d < D_; ++d) {
    pv[d] *= isum;
    depth += pv[d]*depthv[((size_t)b*D_ + d)*HW_ + hw];
    di    += pv[d]*(float)d;
  }
  int didx = (int)di;                       // astype(int32) truncation; di >= 0
  didx = didx < 0 ? 0 : (didx > D_-1 ? D_-1 : didx);
  float c4 = 0.f;
  for (int j = didx-1; j <= didx+2; ++j) if (j >= 0 && j < D_) c4 += pv[j];
  dout[idx] = depth;
  conf[idx] = c4;
}

// ---------------------------------------------------------------- launch
extern "C" void kernel_launch(void* const* d_in, const int* in_sizes, int n_in,
                              void* d_out, int out_size, void* d_ws, size_t ws_size,
                              hipStream_t stream) {
  (void)in_sizes; (void)n_in; (void)out_size; (void)ws_size;
  const float* feat   = (const float*)d_in[0];
  const float* pm     = (const float*)d_in[1];
  const float* depthv = (const float*)d_in[2];
  const float* rw1 = (const float*)d_in[3];
  const float* rb1 = (const float*)d_in[4];
  const float* rw2 = (const float*)d_in[5];
  const float* rb2 = (const float*)d_in[6];
  const float* pw1 = (const float*)d_in[7];
  const float* pb1 = (const float*)d_in[8];
  const float* pw2 = (const float*)d_in[9];
  const float* pb2 = (const float*)d_in[10];
  const float* pw3 = (const float*)d_in[11];
  const float* pb3 = (const float*)d_in[12];
  float* out = (float*)d_out;
  float* ws  = (float*)d_ws;
  float* wsproj = ws + WS_PROJ;
  float* big    = ws + WS_BIG;   // sim_tmp during view loop, conv1 activations after
  float* wsum   = ws + WS_WSUM;

  setup_proj_kernel<<<1, 32, 0, stream>>>(pm, wsproj);

  const long NWARP = (long)B_*D_*HW_;
  const long NSIM  = (long)B_*G_*D_*HW_;
  const long NPIX  = (long)B_*HW_;

  for (int v = 1; v < V_; ++v) {
    warp_sim_kernel<<<(int)((NWARP+255)/256), 256, 0, stream>>>(feat, depthv, wsproj, big, v);
    view_weight_kernel<<<(int)((NPIX+255)/256), 256, 0, stream>>>(
        big, pw1, pb1, pw2, pb2, pw3, pb3, out + OUT_VW, wsum, v, v == 1 ? 1 : 0);
    accum_sim_kernel<<<(int)((NSIM+255)/256), 256, 0, stream>>>(
        big, out + OUT_VW, out + OUT_SIM, v, v == 1 ? 1 : 0);
  }
  norm_sim_kernel<<<(int)((NSIM+255)/256), 256, 0, stream>>>(out + OUT_SIM, wsum);
  conv1_wmma_kernel<<<CV1_BLOCKS, CV1_THREADS, 0, stream>>>(out + OUT_SIM, rw1, rb1, big);
  conv2_kernel<<<(int)((NWARP+255)/256), 256, 0, stream>>>(big, rw2, rb2, out + OUT_PVP);
  depth_kernel<<<(int)((NPIX+255)/256), 256, 0, stream>>>(
      out + OUT_PVP, depthv, out + OUT_DEPTH, out + OUT_CONF);
}
